// PhysBiformer_32564442038684
// MI455X (gfx1250) — matrix operand order repaired and checked by reading, not compiled, not run on hardware
//
#include <hip/hip_runtime.h>
#include <hip/hip_bf16.h>
#include <math.h>

typedef _Float16 half_t;
typedef __attribute__((ext_vector_type(16))) _Float16 v16h;
typedef __attribute__((ext_vector_type(8)))  float    v8f;

// ---------------- constants ----------------
#define C_DIM   64
#define TSTEP   4
#define LT      40
#define LH      16
#define LW      16
#define SPAT    (LT*LH*LW)          // 10240
#define NTOK    (TSTEP*SPAT)        // 40960
#define RWIN    32                  // WT*WH*WW
#define SREG    320                 // rt*rh*rw
#define NKEY    640                 // TOPK*SREG

// ================= helpers: WMMA tile GEMM =================
// A: row-major f16 [M,K]. W: weight in [O,C] (= [N,K]) row-major f16, so B[k][n] = W[n][k].
// CDNA5 16x16x32 f16 layouts:
//  A: row = lane&15; halves 0..7 -> K = kb+h, halves 8..15 -> K = kb+16+(h-8), kb = (lane>>4)*8
//  B: col = lane&15; halves 0..7 -> K = kb+h, halves 8..15 -> K = kb+16+(h-8), kb = (lane>>4)*8
//  C/D: elem i -> row = (lane>>4)*8 + i, col = lane&15
__device__ inline v8f gemm_tile(const half_t* __restrict__ A, int lda,
                                const half_t* __restrict__ W, int ldw,
                                int K, int row0, int col0, int lane) {
  v8f acc = {};
  for (int k0 = 0; k0 < K; k0 += 32) {
    v16h a;
    {
      int r  = row0 + (lane & 15);
      int kb = k0 + ((lane >> 4) << 3);
      const half_t* p = A + (size_t)r * lda + kb;
#pragma unroll
      for (int t = 0; t < 8; ++t)  a[t] = p[t];
#pragma unroll
      for (int t = 8; t < 16; ++t) a[t] = p[t + 8];
    }
    v16h bm;
    {
      int n  = col0 + (lane & 15);
      int kb = k0 + ((lane >> 4) << 3);
      const half_t* p = W + (size_t)n * ldw + kb;
#pragma unroll
      for (int t = 0; t < 16; ++t) bm[t] = p[t + ((t >= 8) ? 8 : 0)];
    }
    acc = __builtin_amdgcn_wmma_f32_16x16x32_f16(false, a, false, bm, (short)0, acc, false, false);
  }
  return acc;
}

// ================= small utility kernels =================
__global__ void k_f2h(const float* __restrict__ s, half_t* __restrict__ d, int n) {
  int i = blockIdx.x * blockDim.x + threadIdx.x;
  if (i < n) d[i] = (half_t)s[i];
}

// per-channel batch stats: element(c,t,sp) = src[(t*C+c)*SP+sp]; stat[c]=mean, stat[C+c]=rsqrt(var+eps)
__global__ __launch_bounds__(256) void k_stats(const float* __restrict__ src, int C, int T, int SP,
                                               float* __restrict__ stat) {
  __shared__ float s1[256], s2[256];
  int c = blockIdx.x, tid = threadIdx.x;
  int N = T * SP;
  float a = 0.f, b = 0.f;
  for (int i = tid; i < N; i += 256) {
    int t = i / SP, sp = i - t * SP;
    float v = src[((size_t)(t * C + c)) * SP + sp];
    a += v; b += v * v;
  }
  s1[tid] = a; s2[tid] = b; __syncthreads();
  for (int o = 128; o > 0; o >>= 1) {
    if (tid < o) { s1[tid] += s1[tid + o]; s2[tid] += s2[tid + o]; }
    __syncthreads();
  }
  if (tid == 0) {
    float m = s1[0] / N;
    float var = s2[0] / N - m * m;
    stat[c] = m;
    stat[C + c] = rsqrtf(var + 1e-5f);
  }
}

// BN apply + activation in place. act: 0=ReLU, 1=ELU
__global__ void k_bnact(float* __restrict__ buf, const float* __restrict__ stat,
                        const float* __restrict__ g, const float* __restrict__ be,
                        int C, int SP, int act) {
  int gid = blockIdx.x * blockDim.x + threadIdx.x;
  if (gid >= C * SP) return;
  int c = gid / SP;
  float v = (buf[gid] - stat[c]) * stat[C + c] * g[c] + be[c];
  buf[gid] = (act == 0) ? fmaxf(v, 0.f) : (v > 0.f ? v : expf(v) - 1.f);
}

// ================= stem convolutions =================
__global__ void k_stem1(const float* __restrict__ x, const float* __restrict__ w,
                        const float* __restrict__ b, float* __restrict__ out) {
  int gid = blockIdx.x * blockDim.x + threadIdx.x;
  if (gid >= 16 * SPAT) return;
  int co = gid / SPAT, sp = gid - co * SPAT;
  int od = sp / 256, oh = (sp >> 4) & 15, ow = sp & 15;
  float acc = b[co];
  for (int ci = 0; ci < 3; ++ci)
    for (int kd = 0; kd < 3; ++kd) {
      int id = od * 4 - 1 + kd; if ((unsigned)id >= 160u) continue;
      for (int kh = 0; kh < 3; ++kh) {
        int ih = oh * 4 - 1 + kh; if ((unsigned)ih >= 64u) continue;
        for (int kw = 0; kw < 3; ++kw) {
          int iw = ow * 4 - 1 + kw; if ((unsigned)iw >= 64u) continue;
          acc += w[(((co * 3 + ci) * 3 + kd) * 3 + kh) * 3 + kw] *
                 x[((ci * 160 + id) * 64 + ih) * 64 + iw];
        }
      }
    }
  out[gid] = acc;
}

__global__ void k_stem2(const float* __restrict__ in, const float* __restrict__ w,
                        const float* __restrict__ b, float* __restrict__ out) {
  int gid = blockIdx.x * blockDim.x + threadIdx.x;
  if (gid >= C_DIM * SPAT) return;
  int co = gid / SPAT, sp = gid - co * SPAT;
  int od = sp / 256, oh = (sp >> 4) & 15, ow = sp & 15;
  float acc = b[co];
  for (int ci = 0; ci < 16; ++ci)
    for (int kd = 0; kd < 3; ++kd) {
      int id = od - 1 + kd; if ((unsigned)id >= (unsigned)LT) continue;
      for (int kh = 0; kh < 3; ++kh) {
        int ih = oh - 1 + kh; if ((unsigned)ih >= (unsigned)LH) continue;
        for (int kw = 0; kw < 3; ++kw) {
          int iw = ow - 1 + kw; if ((unsigned)iw >= (unsigned)LW) continue;
          acc += w[(((co * 16 + ci) * 3 + kd) * 3 + kh) * 3 + kw] *
                 in[((ci * LT + id) * LH + ih) * LW + iw];
        }
      }
    }
  out[gid] = acc;
}

// BN + ReLU + pos_embed, broadcast to T time steps into X[t][c][sp]
__global__ void k_stem2_apply(const float* __restrict__ src, const float* __restrict__ stat,
                              const float* __restrict__ g, const float* __restrict__ be,
                              const float* __restrict__ pos, float* __restrict__ X) {
  int gid = blockIdx.x * blockDim.x + threadIdx.x;
  if (gid >= C_DIM * SPAT) return;
  int c = gid / SPAT, sp = gid - c * SPAT;
  int lt = sp / 256;
  float v = (src[gid] - stat[c]) * stat[C_DIM + c] * g[c] + be[c];
  v = fmaxf(v, 0.f) + pos[c * LT + lt];
  for (int t = 0; t < TSTEP; ++t)
    X[((size_t)(t * C_DIM + c)) * SPAT + sp] = v;
}

// ================= LIF (BN fused), writes f16 spikes =================
// mode 0: attention layout [(t*32+r)*320+s][c];  mode 1: channel-last [(t*SPAT+sp)][c]
__global__ void k_lif(const float* __restrict__ X, const float* __restrict__ stat,
                      const float* __restrict__ g, const float* __restrict__ b,
                      half_t* __restrict__ out, int mode) {
  int gid = blockIdx.x * blockDim.x + threadIdx.x;
  if (gid >= C_DIM * SPAT) return;
  int c = gid / SPAT, sp = gid - c * SPAT;
  float m = stat[c], inv = stat[C_DIM + c], gg = g[c], bb = b[c];
  size_t oidx;
  if (mode == 0) {
    int lt = sp / 256, lh = (sp >> 4) & 15, lw = sp & 15;
    int wt = lt / 20, it = lt - wt * 20;
    int wh = lh >> 2, ih = lh & 3;
    int ww = lw >> 2, iw = lw & 3;
    int r = (wt * 4 + wh) * 4 + ww;
    int s = (it * 4 + ih) * 4 + iw;
    oidx = ((size_t)(r * SREG + s)) * C_DIM + c;   // add t*RWIN*SREG*C below
  } else {
    oidx = ((size_t)sp) * C_DIM + c;
  }
  size_t tstride = (mode == 0) ? (size_t)RWIN * SREG * C_DIM : (size_t)SPAT * C_DIM;
  float v = 0.f;
  for (int t = 0; t < TSTEP; ++t) {
    float xn = (X[((size_t)(t * C_DIM + c)) * SPAT + sp] - m) * inv * gg + bb;
    v = 0.5f * (v + xn);                     // v += (x - v)/tau, tau=2
    float s = (v >= 1.0f) ? 1.0f : 0.0f;     // spike(v - vth), vth=1
    out[oidx + (size_t)t * tstride] = (half_t)s;
    v -= s;                                   // soft reset
  }
}

// ================= GEMM kernels (WMMA) =================
__global__ __launch_bounds__(128) void k_gemm_qkv(const half_t* __restrict__ A,
                                                  const half_t* __restrict__ W,
                                                  const float* __restrict__ bias,
                                                  half_t* __restrict__ Q, half_t* __restrict__ K,
                                                  half_t* __restrict__ V) {
  int warp = threadIdx.x >> 5, lane = threadIdx.x & 31;
  int tile = blockIdx.x * 4 + warp;
  int nt = tile % 12, mt = tile / 12;                 // N=192, M=40960
  v8f acc = gemm_tile(A, 64, W, 64, 64, mt * 16, nt * 16, lane);
  int col = nt * 16 + (lane & 15);
  float bz = bias[col];
  half_t* dst = (col < 64) ? Q : (col < 128 ? K : V);
  int c = col & 63;
  int rb = (lane >> 4) * 8;
#pragma unroll
  for (int i = 0; i < 8; ++i) {
    int m = mt * 16 + rb + i;
    dst[(size_t)m * 64 + c] = (half_t)(acc[i] + bz);
  }
}

__global__ __launch_bounds__(128) void k_gemm_proj(const half_t* __restrict__ A,
                                                   const half_t* __restrict__ W,
                                                   const float* __restrict__ bias,
                                                   const float* __restrict__ scale, int blk,
                                                   float* __restrict__ X) {
  int warp = threadIdx.x >> 5, lane = threadIdx.x & 31;
  int tile = blockIdx.x * 4 + warp;
  int nt = tile % 4, mt = tile / 4;                   // N=64
  v8f acc = gemm_tile(A, 64, W, 64, 64, mt * 16, nt * 16, lane);
  float sc = scale[blk];
  int c = nt * 16 + (lane & 15);
  float bz = bias[c];
  int rb = (lane >> 4) * 8;
#pragma unroll
  for (int i = 0; i < 8; ++i) {
    int m = mt * 16 + rb + i;
    int tb = m / (RWIN * SREG);
    int rr = (m / SREG) & 31;
    int s = m % SREG;
    int wt = rr >> 4, wh = (rr >> 2) & 3, ww = rr & 3;
    int it = s >> 4,  ih = (s >> 2) & 3,  iw = s & 3;
    int lt = wt * 20 + it, lh = wh * 4 + ih, lw = ww * 4 + iw;
    int sp = (lt * LH + lh) * LW + lw;
    X[((size_t)(tb * C_DIM + c)) * SPAT + sp] += (acc[i] + bz) * sc;
  }
}

__global__ __launch_bounds__(128) void k_gemm_ffn1(const half_t* __restrict__ A,
                                                   const half_t* __restrict__ W,
                                                   const float* __restrict__ bias,
                                                   half_t* __restrict__ H) {
  int warp = threadIdx.x >> 5, lane = threadIdx.x & 31;
  int tile = blockIdx.x * 4 + warp;
  int nt = tile % 16, mt = tile / 16;                 // N=256
  v8f acc = gemm_tile(A, 64, W, 64, 64, mt * 16, nt * 16, lane);
  int c = nt * 16 + (lane & 15);
  float bz = bias[c];
  int rb = (lane >> 4) * 8;
#pragma unroll
  for (int i = 0; i < 8; ++i) {
    int m = mt * 16 + rb + i;
    float v = acc[i] + bz;
    float ge = 0.5f * v * (1.0f + erff(v * 0.70710678118f));   // exact GELU
    H[(size_t)m * 256 + c] = (half_t)ge;
  }
}

__global__ __launch_bounds__(128) void k_gemm_ffn2(const half_t* __restrict__ A,
                                                   const half_t* __restrict__ W,
                                                   const float* __restrict__ bias,
                                                   const float* __restrict__ scale, int blk,
                                                   float* __restrict__ X) {
  int warp = threadIdx.x >> 5, lane = threadIdx.x & 31;
  int tile = blockIdx.x * 4 + warp;
  int nt = tile % 4, mt = tile / 4;                   // N=64, K=256
  v8f acc = gemm_tile(A, 256, W, 256, 256, mt * 16, nt * 16, lane);
  float sc = scale[blk];
  int c = nt * 16 + (lane & 15);
  float bz = bias[c];
  int rb = (lane >> 4) * 8;
#pragma unroll
  for (int i = 0; i < 8; ++i) {
    int m = mt * 16 + rb + i;
    int t = m / SPAT, sp = m - t * SPAT;
    X[((size_t)(t * C_DIM + c)) * SPAT + sp] += (acc[i] + bz) * sc;
  }
}

// ================= routing: region means + top-2 =================
__global__ __launch_bounds__(64) void k_regmean(const half_t* __restrict__ Q,
                                                const half_t* __restrict__ K,
                                                float* __restrict__ QM, float* __restrict__ KM) {
  int br = blockIdx.x;                  // tb*32 + r
  int c = threadIdx.x;
  const half_t* q = Q + (size_t)br * SREG * 64 + c;
  const half_t* k = K + (size_t)br * SREG * 64 + c;
  float sq = 0.f, sk = 0.f;
  for (int s = 0; s < SREG; ++s) { sq += (float)q[s * 64]; sk += (float)k[s * 64]; }
  QM[br * 64 + c] = sq * (1.0f / SREG);
  KM[br * 64 + c] = sk * (1.0f / SREG);
}

__global__ __launch_bounds__(32) void k_top2(const float* __restrict__ QM,
                                             const float* __restrict__ KM,
                                             int* __restrict__ IDX) {
  int b = blockIdx.x;                  // tb*32 + r
  int tb = b >> 5;
  int lane = threadIdx.x;              // lane == candidate region (S == 32)
  const float* qm = QM + (size_t)b * 64;
  const float* km = KM + ((size_t)(tb * 32 + lane)) * 64;
  float aff = 0.f;
  for (int c = 0; c < 64; ++c) aff += qm[c] * km[c];
  // argmax (tie-break: lowest index) butterfly
  float v = aff; int id = lane;
  for (int o = 16; o; o >>= 1) {
    float ov = __shfl_xor(v, o, 32);
    int oi = __shfl_xor(id, o, 32);
    if (ov > v || (ov == v && oi < id)) { v = ov; id = oi; }
  }
  int w0 = id;
  float v2 = (lane == w0) ? -3.0e38f : aff; int id2 = lane;
  for (int o = 16; o; o >>= 1) {
    float ov = __shfl_xor(v2, o, 32);
    int oi = __shfl_xor(id2, o, 32);
    if (ov > v2 || (ov == v2 && oi < id2)) { v2 = ov; id2 = oi; }
  }
  if (lane == 0) { IDX[b * 2] = w0; IDX[b * 2 + 1] = id2; }
}

// ================= attention core (one wave per tb,r,h,q-tile) =================
__global__ __launch_bounds__(32) void k_attn(const half_t* __restrict__ Q,
                                             const half_t* __restrict__ K,
                                             const half_t* __restrict__ V,
                                             const int* __restrict__ IDX,
                                             half_t* __restrict__ O) {
  __shared__ float sc[16 * NKEY];                     // 40 KB
  int b = blockIdx.x;
  int qt = b % 20;
  int h = (b / 20) & 3;
  int r = (b / 80) & 31;
  int tb = b / 2560;
  int lane = threadIdx.x;
  int i0 = IDX[(tb * 32 + r) * 2 + 0];
  int i1 = IDX[(tb * 32 + r) * 2 + 1];

  // A tile: q rows [qt*16, qt*16+16), K = dh(16) zero-padded to 32
  v16h a;
  {
    int row = qt * 16 + (lane & 15);
    int kb = (lane >> 4) << 3;
    const half_t* qp = Q + (((size_t)(tb * 32 + r) * SREG + row)) * 64 + h * 16;
#pragma unroll
    for (int t = 0; t < 8; ++t) { int kk = kb + t;     a[t] = (kk < 16) ? qp[kk] : (_Float16)0.0f; }
#pragma unroll
    for (int t = 8; t < 16; ++t) { int kk = kb + t + 8; a[t] = (kk < 16) ? qp[kk] : (_Float16)0.0f; }
  }

  // scores [16,640] into LDS
  for (int kt = 0; kt < NKEY / 16; ++kt) {
    v16h bm;
    {
      int n = kt * 16 + (lane & 15);                 // global key 0..639
      int j = (n >= SREG) ? 1 : 0;
      int srcr = j ? i1 : i0;
      int s = n - j * SREG;
      const half_t* kp = K + (((size_t)(tb * 32 + srcr) * SREG + s)) * 64 + h * 16;
      int kb = (lane >> 4) << 3;
#pragma unroll
      for (int t = 0; t < 16; ++t) { int kk = kb + t + ((t >= 8) ? 8 : 0); bm[t] = (kk < 16) ? kp[kk] : (_Float16)0.0f; }
    }
    v8f accs = {};
    accs = __builtin_amdgcn_wmma_f32_16x16x32_f16(false, a, false, bm, (short)0, accs, false, false);
    int col = kt * 16 + (lane & 15);
    int rb = (lane >> 4) * 8;
#pragma unroll
    for (int i = 0; i < 8; ++i) sc[(rb + i) * NKEY + col] = accs[i] * 0.25f;  // 1/sqrt(16)
  }
  __syncthreads();

  // softmax: row = lane&15, each half-wave covers 320 keys
  {
    int row = lane & 15, hf = lane >> 4;
    float* sr = sc + row * NKEY + hf * SREG;
    float m = -3.0e38f;
    for (int i = 0; i < SREG; ++i) m = fmaxf(m, sr[i]);
    m = fmaxf(m, __shfl_xor(m, 16, 32));
    float sum = 0.f;
    for (int i = 0; i < SREG; ++i) { float e = expf(sr[i] - m); sr[i] = e; sum += e; }
    sum += __shfl_xor(sum, 16, 32);
    float inv = 1.0f / sum;
    for (int i = 0; i < SREG; ++i) sr[i] *= inv;
  }
  __syncthreads();

  // o = P[16,640] @ Vg[640,16]
  v8f acco = {};
  for (int k0 = 0; k0 < NKEY; k0 += 32) {
    v16h pa;
    {
      int prow = lane & 15;
      int kb = k0 + ((lane >> 4) << 3);
      const float* pr = sc + prow * NKEY;
#pragma unroll
      for (int t = 0; t < 8; ++t)  pa[t] = (half_t)pr[kb + t];
#pragma unroll
      for (int t = 8; t < 16; ++t) pa[t] = (half_t)pr[kb + t + 8];
    }
    v16h vb;
    {
      int n = lane & 15;                             // dh column
      int kbb = k0 + ((lane >> 4) << 3);
#pragma unroll
      for (int t = 0; t < 16; ++t) {
        int kk = kbb + t + ((t >= 8) ? 8 : 0);
        int j = (kk >= SREG) ? 1 : 0;
        int srcr = j ? i1 : i0;
        int s = kk - j * SREG;
        vb[t] = V[(((size_t)(tb * 32 + srcr) * SREG + s)) * 64 + h * 16 + n];
      }
    }
    acco = __builtin_amdgcn_wmma_f32_16x16x32_f16(false, pa, false, vb, (short)0, acco, false, false);
  }
  {
    int col = lane & 15;
    int rb = (lane >> 4) * 8;
#pragma unroll
    for (int i = 0; i < 8; ++i) {
      int rowq = qt * 16 + rb + i;
      O[(((size_t)(tb * 32 + r) * SREG + rowq)) * 64 + h * 16 + col] = (half_t)acco[i];
    }
  }
}

// ================= tail: temporal mean, upsample convs, head =================
__global__ void k_meanT(const float* __restrict__ X, float* __restrict__ Y) {
  int gid = blockIdx.x * blockDim.x + threadIdx.x;
  if (gid >= C_DIM * SPAT) return;
  float s = 0.f;
  for (int t = 0; t < TSTEP; ++t) s += X[(size_t)t * C_DIM * SPAT + gid];
  Y[gid] = s * 0.25f;
}

__global__ void k_up1(const float* __restrict__ Y, const float* __restrict__ w,
                      const float* __restrict__ b, float* __restrict__ out) {
  int gid = blockIdx.x * blockDim.x + threadIdx.x;
  if (gid >= 64 * 80 * 256) return;
  int co = gid / (80 * 256);
  int rest = gid - co * 80 * 256;
  int d = rest / 256, hw = rest - d * 256;
  float acc = b[co];
  for (int ci = 0; ci < 64; ++ci)
    for (int kd = 0; kd < 3; ++kd) {
      int dd = d - 1 + kd;
      if ((unsigned)dd >= 80u) continue;
      acc += w[(co * 64 + ci) * 3 + kd] * Y[ci * SPAT + (dd >> 1) * 256 + hw];
    }
  out[gid] = acc;
}

__global__ void k_up2(const float* __restrict__ U, const float* __restrict__ w,
                      const float* __restrict__ b, float* __restrict__ out) {
  int gid = blockIdx.x * blockDim.x + threadIdx.x;
  if (gid >= 32 * 160 * 256) return;
  int co = gid / (160 * 256);
  int rest = gid - co * 160 * 256;
  int d = rest / 256, hw = rest - d * 256;
  float acc = b[co];
  for (int ci = 0; ci < 64; ++ci)
    for (int kd = 0; kd < 3; ++kd) {
      int dd = d - 1 + kd;
      if ((unsigned)dd >= 160u) continue;
      acc += w[(co * 64 + ci) * 3 + kd] * U[ci * 80 * 256 + (dd >> 1) * 256 + hw];
    }
  out[gid] = acc;
}

__global__ void k_ym2(const float* __restrict__ U, float* __restrict__ Y2) {
  int gid = blockIdx.x * blockDim.x + threadIdx.x;
  if (gid >= 32 * 160) return;
  int c = gid / 160, d = gid - c * 160;
  float s = 0.f;
  for (int hw = 0; hw < 256; ++hw) s += U[c * 160 * 256 + d * 256 + hw];
  Y2[gid] = s * (1.0f / 256.0f);
}

__global__ void k_final(const float* __restrict__ Y2, const float* __restrict__ lw,
                        const float* __restrict__ lb, float* __restrict__ out) {
  int d = blockIdx.x * blockDim.x + threadIdx.x;
  if (d >= 160) return;
  float acc = lb[0];
  for (int c = 0; c < 32; ++c) acc += lw[c] * Y2[c * 160 + d];
  out[d] = acc;
}

// ================= host orchestration =================
extern "C" void kernel_launch(void* const* d_in, const int* in_sizes, int n_in,
                              void* d_out, int out_size, void* d_ws, size_t ws_size,
                              hipStream_t stream) {
  const float* x        = (const float*)d_in[0];
  const float* stem_w1  = (const float*)d_in[1];
  const float* stem_b1  = (const float*)d_in[2];
  const float* stem_g1  = (const float*)d_in[3];
  const float* stem_be1 = (const float*)d_in[4];
  const float* stem_w2  = (const float*)d_in[5];
  const float* stem_b2  = (const float*)d_in[6];
  const float* stem_g2  = (const float*)d_in[7];
  const float* stem_be2 = (const float*)d_in[8];
  const float* pos      = (const float*)d_in[9];
  const float* bn1_g    = (const float*)d_in[10];
  const float* bn1_b    = (const float*)d_in[11];
  const float* qkv_w    = (const float*)d_in[12];
  const float* qkv_b    = (const float*)d_in[13];
  const float* proj_w   = (const float*)d_in[14];
  const float* proj_b   = (const float*)d_in[15];
  const float* bn2_g    = (const float*)d_in[16];
  const float* bn2_b    = (const float*)d_in[17];
  const float* ffn_w1   = (const float*)d_in[18];
  const float* ffn_b1   = (const float*)d_in[19];
  const float* ffn_w2   = (const float*)d_in[20];
  const float* ffn_b2   = (const float*)d_in[21];
  const float* blk_sc   = (const float*)d_in[22];
  const float* up1_w    = (const float*)d_in[23];
  const float* up1_b    = (const float*)d_in[24];
  const float* up1_g    = (const float*)d_in[25];
  const float* up1_be   = (const float*)d_in[26];
  const float* up2_w    = (const float*)d_in[27];
  const float* up2_b    = (const float*)d_in[28];
  const float* up2_g    = (const float*)d_in[29];
  const float* up2_be   = (const float*)d_in[30];
  const float* last_w   = (const float*)d_in[31];
  const float* last_b   = (const float*)d_in[32];
  float* out = (float*)d_out;

  char* ws = (char*)d_ws;
  size_t off = 0;
  auto alloc = [&](size_t bytes) -> char* {
    char* p = ws + off;
    off += (bytes + 255) & ~(size_t)255;
    return p;
  };
  float*  X     = (float*)alloc((size_t)TSTEP * C_DIM * SPAT * 4);
  float*  STEM1 = (float*)alloc((size_t)16 * SPAT * 4);
  float*  STEM2 = (float*)alloc((size_t)C_DIM * SPAT * 4);
  float*  STAT  = (float*)alloc(2 * 64 * 4);
  half_t* S1    = (half_t*)alloc((size_t)NTOK * 64 * 2);
  half_t* Qb    = (half_t*)alloc((size_t)NTOK * 64 * 2);
  half_t* Kb    = (half_t*)alloc((size_t)NTOK * 64 * 2);
  half_t* Vb    = (half_t*)alloc((size_t)NTOK * 64 * 2);
  float*  QM    = (float*)alloc((size_t)TSTEP * 32 * 64 * 4);
  float*  KM    = (float*)alloc((size_t)TSTEP * 32 * 64 * 4);
  int*    IDX   = (int*)alloc((size_t)TSTEP * 32 * 2 * 4);
  half_t* OB    = (half_t*)alloc((size_t)NTOK * 64 * 2);
  half_t* S2    = (half_t*)alloc((size_t)NTOK * 64 * 2);
  half_t* Hb    = (half_t*)alloc((size_t)NTOK * 256 * 2);
  half_t* WQ16  = (half_t*)alloc((size_t)3 * 192 * 64 * 2);
  half_t* WP16  = (half_t*)alloc((size_t)3 * 64 * 64 * 2);
  half_t* WF116 = (half_t*)alloc((size_t)3 * 256 * 64 * 2);
  half_t* WF216 = (half_t*)alloc((size_t)3 * 64 * 256 * 2);
  float*  YM    = (float*)alloc((size_t)C_DIM * SPAT * 4);
  float*  U1    = (float*)alloc((size_t)64 * 80 * 256 * 4);
  float*  U2    = (float*)alloc((size_t)32 * 160 * 256 * 4);
  float*  Y2    = (float*)alloc((size_t)32 * 160 * 4);
  (void)ws_size; (void)n_in; (void)in_sizes; (void)out_size;

  // weight conversion fp32 -> f16
  k_f2h<<<(3 * 192 * 64 + 255) / 256, 256, 0, stream>>>(qkv_w, WQ16, 3 * 192 * 64);
  k_f2h<<<(3 * 64 * 64 + 255) / 256, 256, 0, stream>>>(proj_w, WP16, 3 * 64 * 64);
  k_f2h<<<(3 * 256 * 64 + 255) / 256, 256, 0, stream>>>(ffn_w1, WF116, 3 * 256 * 64);
  k_f2h<<<(3 * 64 * 256 + 255) / 256, 256, 0, stream>>>(ffn_w2, WF216, 3 * 64 * 256);

  // stem
  k_stem1<<<(16 * SPAT + 255) / 256, 256, 0, stream>>>(x, stem_w1, stem_b1, STEM1);
  k_stats<<<16, 256, 0, stream>>>(STEM1, 16, 1, SPAT, STAT);
  k_bnact<<<(16 * SPAT + 255) / 256, 256, 0, stream>>>(STEM1, STAT, stem_g1, stem_be1, 16, SPAT, 0);
  k_stem2<<<(C_DIM * SPAT + 255) / 256, 256, 0, stream>>>(STEM1, stem_w2, stem_b2, STEM2);
  k_stats<<<C_DIM, 256, 0, stream>>>(STEM2, C_DIM, 1, SPAT, STAT);
  k_stem2_apply<<<(C_DIM * SPAT + 255) / 256, 256, 0, stream>>>(STEM2, STAT, stem_g2, stem_be2, pos, X);

  // transformer blocks
  for (int blk = 0; blk < 3; ++blk) {
    // BN1 + LIF -> spikes in attention layout
    k_stats<<<C_DIM, 256, 0, stream>>>(X, C_DIM, TSTEP, SPAT, STAT);
    k_lif<<<(C_DIM * SPAT + 255) / 256, 256, 0, stream>>>(X, STAT, bn1_g + blk * 64, bn1_b + blk * 64, S1, 0);
    // QKV GEMM (M=40960, N=192, K=64)
    k_gemm_qkv<<<(2560 * 12) / 4, 128, 0, stream>>>(S1, WQ16 + (size_t)blk * 192 * 64,
                                                    qkv_b + blk * 192, Qb, Kb, Vb);
    // routing
    k_regmean<<<TSTEP * 32, 64, 0, stream>>>(Qb, Kb, QM, KM);
    k_top2<<<TSTEP * 32, 32, 0, stream>>>(QM, KM, IDX);
    // attention
    k_attn<<<TSTEP * 32 * 4 * 20, 32, 0, stream>>>(Qb, Kb, Vb, IDX, OB);
    // proj GEMM + scaled residual scatter
    k_gemm_proj<<<(2560 * 4) / 4, 128, 0, stream>>>(OB, WP16 + (size_t)blk * 64 * 64,
                                                    proj_b + blk * 64, blk_sc, blk, X);
    // BN2 + LIF -> channel-last spikes
    k_stats<<<C_DIM, 256, 0, stream>>>(X, C_DIM, TSTEP, SPAT, STAT);
    k_lif<<<(C_DIM * SPAT + 255) / 256, 256, 0, stream>>>(X, STAT, bn2_g + blk * 64, bn2_b + blk * 64, S2, 1);
    // FFN
    k_gemm_ffn1<<<(2560 * 16) / 4, 128, 0, stream>>>(S2, WF116 + (size_t)blk * 256 * 64,
                                                     ffn_b1 + blk * 256, Hb);
    k_gemm_ffn2<<<(2560 * 4) / 4, 128, 0, stream>>>(Hb, WF216 + (size_t)blk * 64 * 256,
                                                    ffn_b2 + blk * 64, blk_sc, blk, X);
  }

  // temporal mean + upsample head
  k_meanT<<<(C_DIM * SPAT + 255) / 256, 256, 0, stream>>>(X, YM);
  k_up1<<<(64 * 80 * 256 + 255) / 256, 256, 0, stream>>>(YM, up1_w, up1_b, U1);
  k_stats<<<64, 256, 0, stream>>>(U1, 64, 1, 80 * 256, STAT);
  k_bnact<<<(64 * 80 * 256 + 255) / 256, 256, 0, stream>>>(U1, STAT, up1_g, up1_be, 64, 80 * 256, 1);
  k_up2<<<(32 * 160 * 256 + 255) / 256, 256, 0, stream>>>(U1, up2_w, up2_b, U2);
  k_stats<<<32, 256, 0, stream>>>(U2, 32, 1, 160 * 256, STAT);
  k_bnact<<<(32 * 160 * 256 + 255) / 256, 256, 0, stream>>>(U2, STAT, up2_g, up2_be, 32, 160 * 256, 1);
  k_ym2<<<(32 * 160 + 255) / 256, 256, 0, stream>>>(U2, Y2);
  k_final<<<1, 256, 0, stream>>>(Y2, last_w, last_b, out);
}
